// MoE_50663434224155
// MI455X (gfx1250) — matrix operand order repaired and checked by reading, not compiled, not run on hardware
//
#include <hip/hip_runtime.h>
#include <hip/hip_bf16.h>

typedef __attribute__((ext_vector_type(16))) _Float16 v16h;
typedef __attribute__((ext_vector_type(8)))  float    v8f;
typedef int v4i __attribute__((vector_size(16)));

#define NE    8        // experts
#define HD    256      // hidden dim
#define NPTS  65536
#define TM    128      // points per workgroup
#define NHID  3
#define OMEGA 30.0f

// LDS partition (bytes)
#define SW_BYTES   (HD * HD * 2)          // 131072: weight fragments (f16)
#define SA_BYTES   (TM * HD * 2)          //  65536: activation fragments (f16)
#define SG_OFF     (SW_BYTES + SA_BYTES)  // 196608
#define SO_OFF     (SG_OFF + TM * NE * 4) // 200704
#define SX_OFF     (SO_OFF + TM * 3 * 4)  // 202240
#define LDS_TOTAL  (SX_OFF + TM * 2 * 4)  // 203264

#define AS_GLOBAL __attribute__((address_space(1)))
#define AS_LOCAL  __attribute__((address_space(3)))

// Async 16B global->LDS copy (CDNA5 GLOBAL_LOAD_ASYNC_TO_LDS_B128, ASYNCcnt)
__device__ __forceinline__ void async_copy16(void* lds_dst, const void* gsrc) {
#if __has_builtin(__builtin_amdgcn_global_load_async_to_lds_b128)
    __builtin_amdgcn_global_load_async_to_lds_b128(
        (AS_GLOBAL v4i*)gsrc, (AS_LOCAL v4i*)lds_dst, 0, 0);
#else
    unsigned loff = (unsigned)(unsigned long long)(AS_LOCAL void*)lds_dst;
    asm volatile("global_load_async_to_lds_b128 %0, %1, off"
                 :: "v"(loff), "v"(gsrc) : "memory");
#endif
}

__device__ __forceinline__ void async_wait0() {
#if __has_builtin(__builtin_amdgcn_s_wait_asynccnt)
    __builtin_amdgcn_s_wait_asynccnt(0);
#else
    asm volatile("s_wait_asynccnt 0x0" ::: "memory");
#endif
}

// A-fragment address (in halves) for activation element (row, col):
// per 32-wide K block: lanes0-15 hold K {0-7,16-23}, lanes16-31 hold K {8-15,24-31}
__device__ __forceinline__ int addrA(int row, int col) {
    int kb  = col >> 5;
    int kin = col & 31;
    int lh  = (kin >> 3) & 1;                      // which half-lane group
    int i   = (kin & 7) + ((kin >> 4) << 3);       // position within lane's 16 halves
    return ((row << 3) + kb) * 32 + (lh << 4) + i;
}

// Prep: repack Wh fp32 -> f16 B-fragment layout in workspace.
// Tile (nb,kb): 512 halves; lane (0-15 -> col n, K {0-7,16-23}) (16-31 -> col n, K {8-15,24-31}),
// 16 contiguous halves per lane in ascending K.
__global__ __launch_bounds__(256) void moe_prep_wfrag(const float* __restrict__ Wh,
                                                      _Float16* __restrict__ Whf) {
    int idx = blockIdx.x * 256 + threadIdx.x;           // over NE*NHID*HD*HD
    if (idx >= NE * NHID * HD * HD) return;
    int el = idx >> 16;            // e*3 + l
    int kn = idx & 65535;
    int k  = kn >> 8;              // row of W (input feature)
    int n  = kn & 255;             // col of W (output feature)
    int nb = n >> 4, ln = n & 15;
    int kb = k >> 5, kin = k & 31;
    int lane = ln + (((kin >> 3) & 1) << 4);
    int i    = (kin & 7) + ((kin >> 4) << 3);
    Whf[(el << 15) + ((nb * 8 + kb) << 9) + (lane << 4) + i] = (_Float16)Wh[idx];
}

__global__ __launch_bounds__(256) void moe_siren_wmma(
    const float* __restrict__ x,
    const float* __restrict__ gate_W, const float* __restrict__ gate_b,
    const float* __restrict__ W0,     const float* __restrict__ b0,
    const _Float16* __restrict__ Whf, const float* __restrict__ bh,
    const float* __restrict__ Wout,   const float* __restrict__ bout,
    float* __restrict__ out) {

    extern __shared__ char smem[];
    _Float16* sW = (_Float16*)smem;                  // weight fragments
    _Float16* sA = (_Float16*)(smem + SW_BYTES);     // activation fragments
    float*    sG = (float*)(smem + SG_OFF);          // gates [TM][NE]
    float*    sO = (float*)(smem + SO_OFF);          // output accum [TM][3]
    float*    sX = (float*)(smem + SX_OFF);          // coords [TM][2]

    const int t    = threadIdx.x;
    const int lane = t & 31;
    const int wave = t >> 5;          // 0..7
    const int r0   = wave << 4;       // wave's row stripe base
    const int base = blockIdx.x * TM; // global point base

    // ---- gates, coords, output accum init ----
    if (t < TM) {
        float x0 = x[(base + t) * 2 + 0];
        float x1 = x[(base + t) * 2 + 1];
        sX[t * 2 + 0] = x0; sX[t * 2 + 1] = x1;
        float lg[NE], m = -3.0e38f;
        #pragma unroll
        for (int e = 0; e < NE; ++e) {
            lg[e] = x0 * gate_W[e] + x1 * gate_W[NE + e] + gate_b[e];
            m = fmaxf(m, lg[e]);
        }
        float s = 0.f;
        #pragma unroll
        for (int e = 0; e < NE; ++e) { lg[e] = __expf(lg[e] - m); s += lg[e]; }
        float inv = 1.0f / s;
        #pragma unroll
        for (int e = 0; e < NE; ++e) sG[t * NE + e] = lg[e] * inv;
        sO[t * 3 + 0] = 0.f; sO[t * 3 + 1] = 0.f; sO[t * 3 + 2] = 0.f;
    }
    __syncthreads();

    for (int e = 0; e < NE; ++e) {
        // ---- first layer (K=2, scalar) -> sA in A-fragment layout ----
        __syncthreads();   // previous expert's output stage done reading sA
        for (int it = 0; it < 128; ++it) {
            int flat = t + it * 256;
            int r = flat >> 8, c = flat & 255;
            float v = sX[r * 2 + 0] * W0[(e * 2 + 0) * HD + c] +
                      sX[r * 2 + 1] * W0[(e * 2 + 1) * HD + c] + b0[e * HD + c];
            sA[addrA(r, c)] = (_Float16)__sinf(OMEGA * v);
        }

        // ---- hidden layers: WMMA ----
        for (int l = 0; l < NHID; ++l) {
            __syncthreads();  // prior sW readers done (and layer-0 sA writers done)
            {   // async stage of this layer's f16 weight fragments into LDS
                const char* src = (const char*)(Whf + ((size_t)(e * NHID + l) << 15));
                char* dst = (char*)sW;
                #pragma unroll
                for (int i = 0; i < 32; ++i) {
                    int off = (t + i * 256) * 16;
                    async_copy16(dst + off, src + off);
                }
                async_wait0();   // __syncthreads does not cover ASYNCcnt
            }
            __syncthreads();

            // prefetch next weight block (next layer, or next expert's layer 0)
            {
                const _Float16* nxt = nullptr;
                if (l + 1 < NHID)      nxt = Whf + ((size_t)(e * NHID + l + 1) << 15);
                else if (e + 1 < NE)   nxt = Whf + ((size_t)((e + 1) * NHID) << 15);
                if (nxt) {
                    const char* p = (const char*)nxt + t * 512;
                    __builtin_prefetch(p + 0,   0, 1);
                    __builtin_prefetch(p + 128, 0, 1);
                    __builtin_prefetch(p + 256, 0, 1);
                    __builtin_prefetch(p + 384, 0, 1);
                }
            }

            // load this wave's 8 A fragments (rows r0..r0+15, all K)
            v16h a[8];
            const int rowA   = r0 + (lane & 15);
            const int half32 = lane & 16;      // 0 or 16
            #pragma unroll
            for (int kb = 0; kb < 8; ++kb)
                a[kb] = *(const v16h*)(sA + (rowA << 8) + (kb << 5) + half32);

            const int colbase = lane & 15;
            for (int nb = 0; nb < 16; ++nb) {
                int col = (nb << 4) + colbase;
                float bias = bh[(e * NHID + l) * HD + col];
                v8f acc;
                #pragma unroll
                for (int q = 0; q < 8; ++q) acc[q] = bias;   // C seeded with bias

                #pragma unroll
                for (int kb = 0; kb < 8; ++kb) {
                    v16h bfrag = *(const v16h*)(sW + ((nb * 8 + kb) << 9) + (lane << 4));
                    acc = __builtin_amdgcn_wmma_f32_16x16x32_f16(
                              false, a[kb], false, bfrag, (short)0, acc, false, false);
                }

                // sin activation, scatter f16 back to A-fragment layout
                int kb2 = col >> 5, kin = col & 31;
                int lh  = (kin >> 3) & 1;
                int i2  = (kin & 7) + ((kin >> 4) << 3);
                int rowb = r0 + ((lane >> 4) << 3);
                #pragma unroll
                for (int rI = 0; rI < 8; ++rI) {
                    float v = __sinf(OMEGA * acc[rI]);
                    sA[(((rowb + rI) << 3) + kb2) * 32 + (lh << 4) + i2] = (_Float16)v;
                }
            }
        }
        __syncthreads();   // all waves' activations visible

        // ---- output layer (N=3, scalar) + gated accumulate ----
        if (t < TM) {
            float g = sG[t * NE + e];
            #pragma unroll
            for (int o = 0; o < 3; ++o) {
                float sum = bout[e * 3 + o];
                for (int c = 0; c < HD; ++c)
                    sum += (float)sA[addrA(t, c)] * Wout[(e * HD + c) * 3 + o];
                sO[t * 3 + o] += g * sum;
            }
        }
    }
    __syncthreads();

    if (t < TM) {
        out[(base + t) * 3 + 0] = sO[t * 3 + 0];
        out[(base + t) * 3 + 1] = sO[t * 3 + 1];
        out[(base + t) * 3 + 2] = sO[t * 3 + 2];
    }
}

extern "C" void kernel_launch(void* const* d_in, const int* in_sizes, int n_in,
                              void* d_out, int out_size, void* d_ws, size_t ws_size,
                              hipStream_t stream) {
    const float* x      = (const float*)d_in[0];
    const float* gate_W = (const float*)d_in[1];
    const float* gate_b = (const float*)d_in[2];
    const float* W0     = (const float*)d_in[3];
    const float* b0     = (const float*)d_in[4];
    const float* Wh     = (const float*)d_in[5];
    const float* bh     = (const float*)d_in[6];
    const float* Wout   = (const float*)d_in[7];
    const float* bout   = (const float*)d_in[8];
    float* out = (float*)d_out;

    _Float16* Whf = (_Float16*)d_ws;   // 3 MB of f16 weight fragments

    (void)in_sizes; (void)n_in; (void)out_size; (void)ws_size;

    (void)hipFuncSetAttribute((const void*)moe_siren_wmma,
                              hipFuncAttributeMaxDynamicSharedMemorySize, LDS_TOTAL);

    moe_prep_wfrag<<<(NE * NHID * HD * HD + 255) / 256, 256, 0, stream>>>(Wh, Whf);
    moe_siren_wmma<<<NPTS / TM, 256, LDS_TOTAL, stream>>>(
        x, gate_W, gate_b, W0, b0, Whf, bh, Wout, bout, out);
}
